// TeethAlignmentModel_88459146428620
// MI455X (gfx1250) — matrix operand order repaired and checked by reading, not compiled, not run on hardware
//
#include <hip/hip_runtime.h>
#include <hip/hip_bf16.h>

typedef __attribute__((ext_vector_type(16))) __bf16        v16bf;
typedef __attribute__((ext_vector_type(8)))  float         v8f;
typedef __attribute__((ext_vector_type(4)))  unsigned int  v4u;

#define COLS 65536   // B*N = 8*8192 flattened point columns
#define NPTS 8192
#define EPSF 1e-5f

// ---------- helpers ----------
__device__ __forceinline__ unsigned int fenc(float f) {  // order-preserving float->uint
  unsigned int u = __float_as_uint(f);
  return (u & 0x80000000u) ? ~u : (u | 0x80000000u);
}
__device__ __forceinline__ float fdec(unsigned int e) {
  unsigned int u = (e & 0x80000000u) ? (e & 0x7FFFFFFFu) : ~e;
  return __uint_as_float(u);
}
__device__ __forceinline__ unsigned short f2bf(float f) { // round-to-nearest-even bf16
  unsigned int u = __float_as_uint(f);
  u += 0x7FFFu + ((u >> 16) & 1u);
  return (unsigned short)(u >> 16);
}
__device__ __forceinline__ v16bf ldb32(const unsigned short* p) { // 32B -> 16 bf16
  union { v4u q[2]; v16bf v; } u;
  u.q[0] = *reinterpret_cast<const v4u*>(p);
  u.q[1] = *reinterpret_cast<const v4u*>(p + 8);
  return u.v;
}

// ---------- init accumulators ----------
__global__ void k_init(unsigned int* maxenc, float* sum3, float* sq3, float* dist) {
  int i = blockIdx.x * blockDim.x + threadIdx.x;
  if (i < 8192) maxenc[i] = 0x007FFFFFu;      // fenc(-inf)
  if (i < 1024) { sum3[i] = 0.f; sq3[i] = 0.f; }
  if (i < 72)   dist[i] = 0.f;
}

__global__ void k_f32_to_bf16(const float* __restrict__ in, unsigned short* __restrict__ out, int n) {
  int i = blockIdx.x * blockDim.x + threadIdx.x;
  if (i < n) out[i] = f2bf(in[i]);
}

// ---------- conv1: 3 -> 64 (tiny K, plain VALU). y1t layout [col][64] f32 ----------
__global__ void k_conv1(const float* __restrict__ src, const float* __restrict__ w1,
                        const float* __restrict__ b1, float* __restrict__ y1t) {
  int col = blockIdx.x * blockDim.x + threadIdx.x;
  int b = col >> 13, n = col & 8191;
  const float* sb = src + b * 3 * NPTS + n;
  float s0 = sb[0], s1 = sb[NPTS], s2 = sb[2 * NPTS];
  float* yo = y1t + (size_t)col * 64;
#pragma unroll 4
  for (int c = 0; c < 64; ++c)
    yo[c] = fmaf(w1[c * 3 + 0], s0, fmaf(w1[c * 3 + 1], s1, fmaf(w1[c * 3 + 2], s2, b1[c])));
}

// ---------- per-channel sum/sumsq over [COLS][C] f32, one block per channel ----------
__global__ void k_stats(const float* __restrict__ y, int C, float* __restrict__ sum,
                        float* __restrict__ sq) {
  int c = blockIdx.x;
  float s = 0.f, q = 0.f;
  for (int col = threadIdx.x; col < COLS; col += blockDim.x) {
    float v = y[(size_t)col * C + c];
    s += v; q += v * v;
  }
  __shared__ float ss[256], qq[256];
  ss[threadIdx.x] = s; qq[threadIdx.x] = q;
  __syncthreads();
  for (int st = 128; st > 0; st >>= 1) {
    if (threadIdx.x < st) { ss[threadIdx.x] += ss[threadIdx.x + st]; qq[threadIdx.x] += qq[threadIdx.x + st]; }
    __syncthreads();
  }
  if (threadIdx.x == 0) { sum[c] = ss[0]; sq[c] = qq[0]; }
}

// ---------- BN fold: scale = g*rsqrt(var+eps), shift = be - mean*scale ----------
__global__ void k_finalize(const float* sum, const float* sq, const float* g, const float* be,
                           float* scale, float* shift, int C, float invcnt) {
  int c = blockIdx.x * blockDim.x + threadIdx.x;
  if (c >= C) return;
  float m = sum[c] * invcnt;
  float v = sq[c] * invcnt - m * m;
  float s = g[c] * rsqrtf(v + EPSF);
  scale[c] = s;
  shift[c] = be[c] - m * s;
}

// ---------- apply BN+ReLU, emit bf16 activations [col][C] ----------
__global__ void k_apply(const float* __restrict__ y, const float* __restrict__ scale,
                        const float* __restrict__ shift, unsigned short* __restrict__ x, int cmask) {
  int i = blockIdx.x * blockDim.x + threadIdx.x;
  int c = i & cmask;
  float v = fmaf(y[i], scale[c], shift[c]);
  x[i] = f2bf(v > 0.f ? v : 0.f);
}

// ---------- conv2: 64 -> 128 via v_wmma_f32_16x16x32_bf16 ----------
// 8 waves/block, wave = m-tile; two n-tiles in flight -> independent WMMAs fill the
// WMMA->WMMA hazard slots instead of v_nops.
__global__ void k_conv2(const unsigned short* __restrict__ x1t,   // [COLS][64] bf16
                        const unsigned short* __restrict__ w2bf,  // [128][64] bf16
                        const float* __restrict__ b2,
                        float* __restrict__ y2t) {                // [COLS][128] f32
  const int wave = threadIdx.x >> 5;          // m-tile 0..7
  const int lane = threadIdx.x & 31;
  const int l15 = lane & 15;
  const int hiA = (lane >> 4) * 16;           // A fragment K sub-block
  const int kloB = (lane >> 4) * 16;          // B fragment K sub-block
  const int mbase = wave * 16;

  // A tiles (K = 64 -> 2 k-steps), contiguous k = 16h + s (matches B)
  const unsigned short* wrow = w2bf + (size_t)(mbase + l15) * 64;
  v16bf a0, a1;
  { union { v4u q[2]; v16bf v; } u;
    u.q[0] = *(const v4u*)(wrow + hiA);        u.q[1] = *(const v4u*)(wrow + 8 + hiA); a0 = u.v;
    u.q[0] = *(const v4u*)(wrow + 32 + hiA);   u.q[1] = *(const v4u*)(wrow + 40 + hiA); a1 = u.v; }

  float bias[8];
#pragma unroll
  for (int r = 0; r < 8; ++r) bias[r] = b2[mbase + ((lane < 16) ? r : 8 + r)];

  for (int t = 0; t < 16; t += 2) {           // 2 n-tiles per iteration
    int col0 = blockIdx.x * 256 + t * 16 + l15;
    int col1 = col0 + 16;
    const unsigned short* xr0 = x1t + (size_t)col0 * 64 + kloB;
    const unsigned short* xr1 = x1t + (size_t)col1 * 64 + kloB;
    v16bf b00 = ldb32(xr0);
    v16bf b10 = ldb32(xr1);
    v16bf b01 = ldb32(xr0 + 32);
    v16bf b11 = ldb32(xr1 + 32);
    v8f c0 = {}, c1 = {};
    c0 = __builtin_amdgcn_wmma_f32_16x16x32_bf16(false, a0, false, b00, (short)0, c0, false, false);
    c1 = __builtin_amdgcn_wmma_f32_16x16x32_bf16(false, a0, false, b10, (short)0, c1, false, false);
    c0 = __builtin_amdgcn_wmma_f32_16x16x32_bf16(false, a1, false, b01, (short)0, c0, false, false);
    c1 = __builtin_amdgcn_wmma_f32_16x16x32_bf16(false, a1, false, b11, (short)0, c1, false, false);
    float* yo0 = y2t + (size_t)col0 * 128 + mbase;
    float* yo1 = y2t + (size_t)col1 * 128 + mbase;
#pragma unroll
    for (int r = 0; r < 8; ++r) {
      int m = (lane < 16) ? r : 8 + r;
      yo0[m] = c0[r] + bias[r];
      yo1[m] = c1[r] + bias[r];
    }
  }
}

// ---------- conv3: 128 -> 1024, fused BN-stats + per-(batch,channel) max; no y3 materialized ----------
// Two n-tiles in flight per wave; WMMAs on c0/c1 interleave so dependent matrix ops never
// issue back-to-back.
__global__ void k_conv3(const unsigned short* __restrict__ x2t,   // [COLS][128] bf16
                        const unsigned short* __restrict__ w3bf,  // [1024][128] bf16
                        const float* __restrict__ b3,
                        float* __restrict__ sum3, float* __restrict__ sq3,
                        unsigned int* __restrict__ maxenc) {      // [8][1024]
  const int gid  = blockIdx.x * 8 + (threadIdx.x >> 5);  // 16384 waves
  const int lane = threadIdx.x & 31;
  const int l15 = lane & 15;
  const int hiA = (lane >> 4) * 16;
  const int kloB = (lane >> 4) * 16;
  const int mtile = gid >> 8;                 // 0..63
  const int chunk = gid & 255;                // 256-column chunk (stays inside one batch)
  const int mbase = mtile * 16;
  const int batch = chunk >> 5;               // 8192 cols / batch, 256 cols / chunk

  const unsigned short* wrow = w3bf + (size_t)(mbase + l15) * 128;
  v16bf a[4];
#pragma unroll
  for (int ks = 0; ks < 4; ++ks) {
    union { v4u q[2]; v16bf v; } u;
    u.q[0] = *(const v4u*)(wrow + ks * 32 + hiA);
    u.q[1] = *(const v4u*)(wrow + ks * 32 + 8 + hiA);
    a[ks] = u.v;
  }
  float bias[8], vmax[8], vsum[8], vsq[8];
#pragma unroll
  for (int r = 0; r < 8; ++r) {
    bias[r] = b3[mbase + ((lane < 16) ? r : 8 + r)];
    vmax[r] = -3.4e38f; vsum[r] = 0.f; vsq[r] = 0.f;
  }

  for (int t = 0; t < 16; t += 2) {           // 2 n-tiles per iteration
    int col0 = chunk * 256 + t * 16 + l15;
    const unsigned short* xr0 = x2t + (size_t)col0 * 128 + kloB;
    const unsigned short* xr1 = xr0 + 16 * 128;   // next 16-column tile
    v8f c0 = {}, c1 = {};
#pragma unroll
    for (int ks = 0; ks < 4; ++ks) {
      v16bf bt0 = ldb32(xr0 + ks * 32);
      v16bf bt1 = ldb32(xr1 + ks * 32);
      c0 = __builtin_amdgcn_wmma_f32_16x16x32_bf16(false, a[ks], false, bt0, (short)0, c0, false, false);
      c1 = __builtin_amdgcn_wmma_f32_16x16x32_bf16(false, a[ks], false, bt1, (short)0, c1, false, false);
    }
#pragma unroll
    for (int r = 0; r < 8; ++r) {
      float v0 = c0[r] + bias[r];
      float v1 = c1[r] + bias[r];
      vmax[r] = fmaxf(vmax[r], fmaxf(v0, v1));
      vsum[r] += v0 + v1;
      vsq[r]  += v0 * v0 + v1 * v1;
    }
  }
  // reduce across the 16 lanes that share each row group
#pragma unroll
  for (int r = 0; r < 8; ++r) {
    float mx = vmax[r], s = vsum[r], q = vsq[r];
    for (int off = 1; off < 16; off <<= 1) {
      mx = fmaxf(mx, __shfl_xor(mx, off, 16));
      s += __shfl_xor(s, off, 16);
      q += __shfl_xor(q, off, 16);
    }
    if (l15 == 0) {
      int m = mbase + ((lane < 16) ? r : 8 + r);
      atomicMax(&maxenc[batch * 1024 + m], fenc(mx));
      atomicAdd(&sum3[m], s);
      atomicAdd(&sq3[m], q);
    }
  }
}

// ---------- head: bn3+relu on pooled maxes, fc1+bn+relu, fc2+bn+relu, rot/trans ----------
__global__ void k_head(const unsigned int* __restrict__ maxenc,
                       const float* __restrict__ sc3, const float* __restrict__ sh3,
                       const float* __restrict__ fw1, const float* __restrict__ fb1,
                       const float* __restrict__ g4, const float* __restrict__ be4,
                       const float* __restrict__ fw2, const float* __restrict__ fb2,
                       const float* __restrict__ g5, const float* __restrict__ be5,
                       const float* __restrict__ wr, const float* __restrict__ br,
                       const float* __restrict__ wt, const float* __restrict__ bt,
                       float* __restrict__ rotb, float* __restrict__ trb) {
  __shared__ float x3s[8 * 1024];
  __shared__ float y4s[8 * 512];
  __shared__ float y5s[8 * 256];
  const int tid = threadIdx.x;

  for (int i = tid; i < 8 * 1024; i += 256) {
    int c = i & 1023;
    float v = fmaf(fdec(maxenc[i]), sc3[c], sh3[c]);
    x3s[i] = v > 0.f ? v : 0.f;
  }
  __syncthreads();
  for (int i = tid; i < 8 * 512; i += 256) {
    int b = i >> 9, o = i & 511;
    const float* w = fw1 + (size_t)o * 1024;
    const float* xr = x3s + b * 1024;
    float acc = fb1[o];
    for (int k = 0; k < 1024; ++k) acc = fmaf(xr[k], w[k], acc);
    y4s[i] = acc;
  }
  __syncthreads();
  for (int o = tid; o < 512; o += 256) {
    float m = 0.f;
    for (int b = 0; b < 8; ++b) m += y4s[b * 512 + o];
    m *= 0.125f;
    float v = 0.f;
    for (int b = 0; b < 8; ++b) { float d = y4s[b * 512 + o] - m; v += d * d; }
    v *= 0.125f;
    float s = g4[o] * rsqrtf(v + EPSF), t = be4[o] - m * s;
    for (int b = 0; b < 8; ++b) {
      float z = fmaf(y4s[b * 512 + o], s, t);
      y4s[b * 512 + o] = z > 0.f ? z : 0.f;
    }
  }
  __syncthreads();
  for (int i = tid; i < 8 * 256; i += 256) {
    int b = i >> 8, o = i & 255;
    const float* w = fw2 + (size_t)o * 512;
    const float* xr = y4s + b * 512;
    float acc = fb2[o];
    for (int k = 0; k < 512; ++k) acc = fmaf(xr[k], w[k], acc);
    y5s[i] = acc;
  }
  __syncthreads();
  if (tid < 256) {
    int o = tid;
    float m = 0.f;
    for (int b = 0; b < 8; ++b) m += y5s[b * 256 + o];
    m *= 0.125f;
    float v = 0.f;
    for (int b = 0; b < 8; ++b) { float d = y5s[b * 256 + o] - m; v += d * d; }
    v *= 0.125f;
    float s = g5[o] * rsqrtf(v + EPSF), t = be5[o] - m * s;
    for (int b = 0; b < 8; ++b) {
      float z = fmaf(y5s[b * 256 + o], s, t);
      y5s[b * 256 + o] = z > 0.f ? z : 0.f;
    }
  }
  __syncthreads();
  if (tid < 72) {
    int b = tid / 9, ij = tid % 9;
    const float* xr = y5s + b * 256;
    const float* w = wr + ij * 256;
    float acc = br[ij] + ((ij == 0 || ij == 4 || ij == 8) ? 1.f : 0.f);  // + eye(3)
    for (int k = 0; k < 256; ++k) acc = fmaf(xr[k], w[k], acc);
    rotb[tid] = acc;
  } else if (tid < 96) {
    int t2 = tid - 72, b = t2 / 3, c = t2 % 3;
    const float* xr = y5s + b * 256;
    const float* w = wt + c * 256;
    float acc = bt[c];
    for (int k = 0; k < 256; ++k) acc = fmaf(xr[k], w[k], acc);
    trb[t2] = acc;
  }
}

// ---------- apply transform, write st, accumulate 3x3 dist matrix per batch ----------
__global__ void k_st(const float* __restrict__ src, const float* __restrict__ tgt,
                     const float* __restrict__ rotb, const float* __restrict__ trb,
                     float* __restrict__ out, float* __restrict__ dist) {
  int col = blockIdx.x * blockDim.x + threadIdx.x;
  int b = col >> 13, n = col & 8191;
  const float* sb = src + b * 3 * NPTS + n;
  float s0 = sb[0], s1 = sb[NPTS], s2 = sb[2 * NPTS];
  const float* R = rotb + b * 9;
  const float* T = trb + b * 3;
  float st[3];
#pragma unroll
  for (int c = 0; c < 3; ++c) {
    st[c] = fmaf(s0, R[c], fmaf(s1, R[3 + c], fmaf(s2, R[6 + c], T[c])));
    out[1 + (size_t)(b * 3 + c) * NPTS + n] = st[c];
  }
  const float* tb = tgt + b * 3 * NPTS + n;
  float tg[3] = { tb[0], tb[NPTS], tb[2 * NPTS] };
  int lane = threadIdx.x & 31;
#pragma unroll
  for (int i = 0; i < 3; ++i)
#pragma unroll
    for (int j = 0; j < 3; ++j) {
      float d = st[j] - tg[i];
      d *= d;
      for (int off = 1; off < 32; off <<= 1) d += __shfl_xor(d, off, 32);
      if (lane == 0) atomicAdd(&dist[b * 9 + i * 3 + j], d);
    }
}

__global__ void k_loss(const float* __restrict__ dist, float* __restrict__ out) {
  if (threadIdx.x != 0 || blockIdx.x != 0) return;
  float acc = 0.f;
  for (int b = 0; b < 8; ++b) {
    const float* d = dist + b * 9;
    for (int j = 0; j < 3; ++j) acc += fminf(d[j], fminf(d[3 + j], d[6 + j]));        // min over i
    for (int i = 0; i < 3; ++i) acc += fminf(d[i * 3], fminf(d[i * 3 + 1], d[i * 3 + 2])); // min over j
  }
  out[0] = acc * (1.0f / 24.0f);
}

// ---------- host ----------
extern "C" void kernel_launch(void* const* d_in, const int* in_sizes, int n_in,
                              void* d_out, int out_size, void* d_ws, size_t ws_size,
                              hipStream_t stream) {
  const float* src = (const float*)d_in[0];
  const float* tgt = (const float*)d_in[1];
  const float* w1  = (const float*)d_in[2];
  const float* b1  = (const float*)d_in[3];
  const float* w2  = (const float*)d_in[4];
  const float* b2  = (const float*)d_in[5];
  const float* w3  = (const float*)d_in[6];
  const float* b3  = (const float*)d_in[7];
  const float* fw1 = (const float*)d_in[8];
  const float* fb1 = (const float*)d_in[9];
  const float* fw2 = (const float*)d_in[10];
  const float* fb2 = (const float*)d_in[11];
  const float* wr  = (const float*)d_in[12];
  const float* br  = (const float*)d_in[13];
  const float* wt  = (const float*)d_in[14];
  const float* bt  = (const float*)d_in[15];
  const float* g1  = (const float*)d_in[16];
  const float* be1 = (const float*)d_in[17];
  const float* g2  = (const float*)d_in[18];
  const float* be2 = (const float*)d_in[19];
  const float* g3  = (const float*)d_in[20];
  const float* be3 = (const float*)d_in[21];
  const float* g4  = (const float*)d_in[22];
  const float* be4 = (const float*)d_in[23];
  const float* g5  = (const float*)d_in[24];
  const float* be5 = (const float*)d_in[25];
  float* out = (float*)d_out;

  char* ws = (char*)d_ws;
  size_t off = 0;
  auto alloc = [&](size_t bytes) -> void* {
    void* p = ws + off;
    off = (off + bytes + 255) & ~(size_t)255;
    return p;
  };
  float*          y1t  = (float*)alloc((size_t)COLS * 64 * 4);
  unsigned short* x1t  = (unsigned short*)alloc((size_t)COLS * 64 * 2);
  float*          y2t  = (float*)alloc((size_t)COLS * 128 * 4);
  unsigned short* x2t  = (unsigned short*)alloc((size_t)COLS * 128 * 2);
  unsigned short* w2bf = (unsigned short*)alloc(128 * 64 * 2);
  unsigned short* w3bf = (unsigned short*)alloc(1024 * 128 * 2);
  float* sum1 = (float*)alloc(64 * 4);    float* sq1 = (float*)alloc(64 * 4);
  float* sum2 = (float*)alloc(128 * 4);   float* sq2 = (float*)alloc(128 * 4);
  float* sum3 = (float*)alloc(1024 * 4);  float* sq3 = (float*)alloc(1024 * 4);
  float* sc1 = (float*)alloc(64 * 4);     float* sh1 = (float*)alloc(64 * 4);
  float* sc2 = (float*)alloc(128 * 4);    float* sh2 = (float*)alloc(128 * 4);
  float* sc3 = (float*)alloc(1024 * 4);   float* sh3 = (float*)alloc(1024 * 4);
  unsigned int* maxenc = (unsigned int*)alloc(8192 * 4);
  float* rotb = (float*)alloc(72 * 4);
  float* trb  = (float*)alloc(24 * 4);
  float* dist = (float*)alloc(72 * 4);
  (void)in_sizes; (void)n_in; (void)out_size; (void)ws_size;

  const float invcnt = 1.0f / (float)COLS;

  k_init<<<32, 256, 0, stream>>>(maxenc, sum3, sq3, dist);
  k_f32_to_bf16<<<(128 * 64 + 255) / 256, 256, 0, stream>>>(w2, w2bf, 128 * 64);
  k_f32_to_bf16<<<(1024 * 128 + 255) / 256, 256, 0, stream>>>(w3, w3bf, 1024 * 128);

  k_conv1<<<COLS / 256, 256, 0, stream>>>(src, w1, b1, y1t);
  k_stats<<<64, 256, 0, stream>>>(y1t, 64, sum1, sq1);
  k_finalize<<<1, 64, 0, stream>>>(sum1, sq1, g1, be1, sc1, sh1, 64, invcnt);
  k_apply<<<COLS * 64 / 256, 256, 0, stream>>>(y1t, sc1, sh1, x1t, 63);

  k_conv2<<<256, 256, 0, stream>>>(x1t, w2bf, b2, y2t);
  k_stats<<<128, 256, 0, stream>>>(y2t, 128, sum2, sq2);
  k_finalize<<<1, 128, 0, stream>>>(sum2, sq2, g2, be2, sc2, sh2, 128, invcnt);
  k_apply<<<COLS * 128 / 256, 256, 0, stream>>>(y2t, sc2, sh2, x2t, 127);

  k_conv3<<<2048, 256, 0, stream>>>(x2t, w3bf, b3, sum3, sq3, maxenc);
  k_finalize<<<1, 1024, 0, stream>>>(sum3, sq3, g3, be3, sc3, sh3, 1024, invcnt);

  k_head<<<1, 256, 0, stream>>>(maxenc, sc3, sh3, fw1, fb1, g4, be4,
                                fw2, fb2, g5, be5, wr, br, wt, bt, rotb, trb);
  k_st<<<COLS / 256, 256, 0, stream>>>(src, tgt, rotb, trb, out, dist);
  k_loss<<<1, 1, 0, stream>>>(dist, out);
}
